// MoE_61040075210969
// MI455X (gfx1250) — matrix operand order repaired and checked by reading, not compiled, not run on hardware
//
#include <hip/hip_runtime.h>
#include <hip/hip_bf16.h>

// MoE dense: B=32, N=2048 -> T=65536 tokens, D_IN=512, E=8, D_EXP=128.
// Main GEMM: X(65536x512) * W^T(512x1024) with bf16x3 split (fp32-class accuracy),
// fused gate softmax + gated expert reduction, all WMMA. Compute-bound.
// Each wave register-blocks M=32 (2 row tiles) so each B fragment feeds 6 WMMAs.

typedef __attribute__((ext_vector_type(16))) __bf16 v16bf;
typedef __attribute__((ext_vector_type(8)))  __bf16 v8bf;
typedef __attribute__((ext_vector_type(8)))  float  v8f;
typedef __attribute__((ext_vector_type(4)))  float  v4f;

#define T_TOK   65536
#define D_IN    512
#define NCOL    1024       // E * D_EXP
#define MT      64         // tokens per workgroup
#define SP      520        // padded LDS row stride (bf16 elems), 1040B = 65*16B

// ---------------------------------------------------------------------------
// Split We (fp32, flat [col=e*128+k][d]) into bf16 hi/lo in WMMA-B packet
// order: packet (kg=d/16, c) holds 16 contiguous K values for column c.
// offset = kg*16384 + c*16 + (d%16)
// ---------------------------------------------------------------------------
__global__ void moe_prep_w(const float* __restrict__ We,
                           __bf16* __restrict__ whi, __bf16* __restrict__ wlo) {
  int idx = blockIdx.x * blockDim.x + threadIdx.x;   // 0 .. 1024*512-1
  int c = idx >> 9;        // column (expert-major)
  int d = idx & 511;       // K index
  float v = We[idx];
  __bf16 h = (__bf16)v;
  __bf16 l = (__bf16)(v - (float)h);
  int off = ((d >> 4) << 14) + (c << 4) + (d & 15);
  whi[off] = h;
  wlo[off] = l;
}

// ---------------------------------------------------------------------------
// Main fused kernel: one workgroup = 64 tokens x all 1024 columns.
// 16 waves: waveM (0..1, 32 tokens each) x waveK (0..7, 16 kout cols each).
// Wave owns 32 tokens x 16 kout cols across all 8 experts -> gated expert
// reduction stays in registers (deterministic, no atomics).
// ---------------------------------------------------------------------------
__global__ __launch_bounds__(512) void moe_main(
    const float*  __restrict__ x,
    const __bf16* __restrict__ whi,
    const __bf16* __restrict__ wlo,
    const float*  __restrict__ be,
    const float*  __restrict__ Wg,
    const float*  __restrict__ bg,
    float*        __restrict__ out,
    float*        __restrict__ gates_out) {

  __shared__ __attribute__((aligned(16))) __bf16 Xhi[MT * SP];
  __shared__ __attribute__((aligned(16))) __bf16 Xlo[MT * SP];
  __shared__ float logit[MT * 8];
  __shared__ float glds[MT * 8];

  const int tid = threadIdx.x;
  const int tokBase = blockIdx.x * MT;

  // ---- Stage X tile (64 rows x 512) to LDS as bf16 hi/lo.
  // float4 loads: 64*512/4 = 8192 vec4 / 512 threads = 16 iterations, each a
  // fully coalesced 8KB sweep (4 rows per iteration).
  for (int i = 0; i < (MT * D_IN / 4) / 512; ++i) {
    int idx4 = i * 512 + tid;
    int base = idx4 * 4;                 // flat element index in the tile
    int r = base >> 9, d = base & 511;
    v4f v = *(const v4f*)(x + (size_t)(tokBase + r) * D_IN + d);
    #pragma unroll
    for (int j = 0; j < 4; ++j) {
      __bf16 h = (__bf16)v[j];
      Xhi[r * SP + d + j] = h;
      Xlo[r * SP + d + j] = (__bf16)(v[j] - (float)h);
    }
  }
  __syncthreads();

  // ---- Gate logits: 64 tokens x 8 experts = 512 threads, one dot each,
  // vectorized by 8 (v8bf LDS loads).
  {
    int t = tid >> 3, e = tid & 7;
    const float*  wg = Wg  + e * D_IN;
    const __bf16* ph = Xhi + t * SP;
    const __bf16* pl = Xlo + t * SP;
    float s = 0.f;
    for (int d = 0; d < D_IN; d += 8) {
      v8bf h = *(const v8bf*)(ph + d);
      v8bf l = *(const v8bf*)(pl + d);
      #pragma unroll
      for (int j = 0; j < 8; ++j)
        s += ((float)h[j] + (float)l[j]) * wg[d + j];
    }
    logit[t * 8 + e] = s + bg[e];
  }
  __syncthreads();

  // ---- Softmax per token; write gate_scores output.
  if (tid < MT) {
    int t = tid;
    float l[8], m = -1e30f;
    for (int e = 0; e < 8; ++e) { l[e] = logit[t * 8 + e]; m = fmaxf(m, l[e]); }
    float s = 0.f;
    for (int e = 0; e < 8; ++e) { l[e] = expf(l[e] - m); s += l[e]; }
    float inv = 1.f / s;
    for (int e = 0; e < 8; ++e) {
      float g = l[e] * inv;
      glds[t * 8 + e] = g;
      gates_out[(size_t)(tokBase + t) * 8 + e] = g;
    }
  }
  __syncthreads();

  // ---- Main WMMA GEMM.
  const int lane  = tid & 31;
  const int w     = tid >> 5;       // 0..15
  const int waveM = w >> 3;         // 32-token half
  const int waveK = w & 7;          // kout block (16 cols)
  const int half  = lane >> 4;
  const int l16   = lane & 15;

  v8f acc0[8] = {};                 // row block 0, one 16x16 tile per expert
  v8f acc1[8] = {};                 // row block 1

  const __bf16* xh0 = Xhi + (waveM * 32 + l16) * SP;
  const __bf16* xl0 = Xlo + (waveM * 32 + l16) * SP;
  const __bf16* xh1 = xh0 + 16 * SP;
  const __bf16* xl1 = xl0 + 16 * SP;

  for (int k0 = 0; k0 < D_IN; k0 += 32) {
    // A fragments: per lane K = koff..koff+15 contiguous (matches B packets),
    // koff = half*16.
    const int ka = k0 + half * 16;
    v16bf ah0, al0, ah1, al1;
    *(v8bf*)&ah0         = *(const v8bf*)(xh0 + ka);
    *(((v8bf*)&ah0) + 1) = *(const v8bf*)(xh0 + ka + 8);
    *(v8bf*)&al0         = *(const v8bf*)(xl0 + ka);
    *(((v8bf*)&al0) + 1) = *(const v8bf*)(xl0 + ka + 8);
    *(v8bf*)&ah1         = *(const v8bf*)(xh1 + ka);
    *(((v8bf*)&ah1) + 1) = *(const v8bf*)(xh1 + ka + 8);
    *(v8bf*)&al1         = *(const v8bf*)(xl1 + ka);
    *(((v8bf*)&al1) + 1) = *(const v8bf*)(xl1 + ka + 8);

    const int kg = (k0 >> 4) + half;   // B fragment K-half by lane>=16
    #pragma unroll
    for (int e = 0; e < 8; ++e) {
      const int c = e * 128 + waveK * 16 + l16;       // column per lane
      const __bf16* bh = whi + (kg << 14) + (c << 4); // packet base
      const __bf16* bl = wlo + (kg << 14) + (c << 4);
      v16bf bhi, blo;
      *(v8bf*)&bhi         = *(const v8bf*)(bh);
      *(((v8bf*)&bhi) + 1) = *(const v8bf*)(bh + 8);
      *(v8bf*)&blo         = *(const v8bf*)(bl);
      *(((v8bf*)&blo) + 1) = *(const v8bf*)(bl + 8);

      // bf16x3 per row block: ah*bh + al*bh + ah*bl, fp32 accumulate.
      // 6 WMMAs per B-fragment pair (M register blocking).
      acc0[e] = __builtin_amdgcn_wmma_f32_16x16x32_bf16(
          false, ah0, false, bhi, (short)0, acc0[e], false, false);
      acc1[e] = __builtin_amdgcn_wmma_f32_16x16x32_bf16(
          false, ah1, false, bhi, (short)0, acc1[e], false, false);
      acc0[e] = __builtin_amdgcn_wmma_f32_16x16x32_bf16(
          false, al0, false, bhi, (short)0, acc0[e], false, false);
      acc1[e] = __builtin_amdgcn_wmma_f32_16x16x32_bf16(
          false, al1, false, bhi, (short)0, acc1[e], false, false);
      acc0[e] = __builtin_amdgcn_wmma_f32_16x16x32_bf16(
          false, ah0, false, blo, (short)0, acc0[e], false, false);
      acc1[e] = __builtin_amdgcn_wmma_f32_16x16x32_bf16(
          false, ah1, false, blo, (short)0, acc1[e], false, false);
    }
  }

  // ---- Epilogue: bias + gated sum over experts, deterministic in-register.
  // C/D layout: VGPR v, lane L -> row = v + (L>=16)*8, col = L&15.
  const int kout = waveK * 16 + l16;
  float bias[8];
  #pragma unroll
  for (int e = 0; e < 8; ++e) bias[e] = be[e * 128 + kout];

  #pragma unroll
  for (int v = 0; v < 8; ++v) {
    const int t0 = waveM * 32 + half * 8 + v;        // row block 0
    const int t1 = t0 + 16;                          // row block 1
    float s0 = 0.f, s1 = 0.f;
    #pragma unroll
    for (int e = 0; e < 8; ++e) {
      s0 += glds[t0 * 8 + e] * (acc0[e][v] + bias[e]);
      s1 += glds[t1 * 8 + e] * (acc1[e][v] + bias[e]);
    }
    out[(size_t)(tokBase + t0) * 128 + kout] = s0;
    out[(size_t)(tokBase + t1) * 128 + kout] = s1;
  }
}

// ---------------------------------------------------------------------------
extern "C" void kernel_launch(void* const* d_in, const int* in_sizes, int n_in,
                              void* d_out, int out_size, void* d_ws, size_t ws_size,
                              hipStream_t stream) {
  const float* x  = (const float*)d_in[0];  // (32,2048,512)
  const float* We = (const float*)d_in[1];  // (8,128,512)
  const float* be = (const float*)d_in[2];  // (8,128)
  const float* Wg = (const float*)d_in[3];  // (8,512)
  const float* bg = (const float*)d_in[4];  // (8,)

  float* out   = (float*)d_out;                       // (T,128)
  float* gates = out + (size_t)T_TOK * 128;           // (T,8)

  __bf16* whi = (__bf16*)d_ws;                        // 1 MB
  __bf16* wlo = whi + (size_t)NCOL * D_IN;            // 1 MB

  moe_prep_w<<<(NCOL * D_IN) / 256, 256, 0, stream>>>(We, whi, wlo);
  moe_main<<<T_TOK / MT, 512, 0, stream>>>(x, whi, wlo, be, Wg, bg, out, gates);
}
